// LinearAttention_25228637896993
// MI455X (gfx1250) — matrix operand order repaired and checked by reading, not compiled, not run on hardware
//
#include <hip/hip_runtime.h>
#include <math.h>

// ---------------------------------------------------------------------------
// Linear attention for MI455X (gfx1250, wave32, WMMA).
// All GEMMs use v_wmma_f32_16x16x32_bf16 (fp32 accumulate).
// Big GEMMs register-blocked 32x64 per wave: 8 WMMAs per 12 b128 loads.
// ---------------------------------------------------------------------------

typedef __attribute__((ext_vector_type(16))) __bf16 bf16x16;
typedef __attribute__((ext_vector_type(8)))  __bf16 bf16x8;
typedef __attribute__((ext_vector_type(8)))  float  f32x8;

#define BB   4
#define SS   8192
#define DMM  512
#define HH   8
#define DD   64
#define MTOT (BB * SS)   // 32768 tokens

__device__ __forceinline__ float elu1(float t) {
    // elu(t) + 1  ==  t>0 ? t+1 : exp(t)
    return t > 0.0f ? t + 1.0f : __expf(t);
}

// A-fragment (16x32 bf16, row-major source): lane hi=0 -> K 0..15,
// lane hi=1 -> K 16..31  (contiguous, matching the B fragments).
__device__ __forceinline__ bf16x16 makeA(const __bf16* __restrict__ row, int k0, int hi) {
    bf16x8 lo  = *(const bf16x8*)(row + k0 + hi * 16);
    bf16x8 hi8 = *(const bf16x8*)(row + k0 + hi * 16 + 8);
    return __builtin_shufflevector(lo, hi8, 0, 1, 2, 3, 4, 5, 6, 7,
                                            8, 9, 10, 11, 12, 13, 14, 15);
}

__device__ __forceinline__ f32x8 wmma_bf16(bf16x16 a, bf16x16 b, f32x8 c) {
    return __builtin_amdgcn_wmma_f32_16x16x32_bf16(false, a, false, b,
                                                   (short)0, c, false, false);
}

// ---------------------------------------------------------------------------
// Prep: fp32 -> bf16 conversion, and weight transpose W[k][n] -> Wt[n][k].
// ---------------------------------------------------------------------------
__global__ void cvt_kernel(const float* __restrict__ X, __bf16* __restrict__ XB, int n) {
    int i = blockIdx.x * blockDim.x + threadIdx.x;
    if (i < n) XB[i] = (__bf16)X[i];
}

__global__ void wT_kernel(const float* __restrict__ W, __bf16* __restrict__ WT) {
    int i = blockIdx.x * blockDim.x + threadIdx.x;      // 512*512 threads
    int k = i >> 9, n = i & 511;
    WT[(size_t)n * DMM + k] = (__bf16)W[(size_t)k * DMM + n];
}

// ---------------------------------------------------------------------------
// QKV projection GEMM: C[32768x512] = XB @ W  (Wt is [N][K] bf16).
// One wave per 32x64 output tile: 2 A-frags x 4 B-frags = 8 WMMAs / K-step.
// MODE 0: q  -> elu+1, store row-major bf16 [token][h*64+d]
// MODE 1: k  -> elu+1, store transposed   [b,h,d,s]  (b128 stores)
// MODE 2: v  ->         store transposed   [b,h,e,s]
// ---------------------------------------------------------------------------
template <int MODE>
__global__ void proj_gemm(const __bf16* __restrict__ X, const __bf16* __restrict__ WT,
                          __bf16* __restrict__ OUT) {
    int wid  = (blockIdx.x * blockDim.x + threadIdx.x) >> 5;  // 8192 waves
    int lane = threadIdx.x & 31;
    int r = lane & 15, hi = lane >> 4;
    int m0 = (wid >> 3) << 5;   // 1024 M tiles of 32
    int n0 = (wid & 7) << 6;    // 8    N tiles of 64

    const __bf16* arow[2];
    const __bf16* brow[4];
#pragma unroll
    for (int i = 0; i < 2; ++i) arow[i] = X + (size_t)(m0 + i * 16 + r) * DMM;
#pragma unroll
    for (int j = 0; j < 4; ++j) brow[j] = WT + (size_t)(n0 + j * 16 + r) * DMM + hi * 16;

    f32x8 acc[2][4] = {};
    for (int k0 = 0; k0 < DMM; k0 += 32) {
        bf16x16 a[2], b[4];
#pragma unroll
        for (int i = 0; i < 2; ++i) a[i] = makeA(arow[i], k0, hi);
#pragma unroll
        for (int j = 0; j < 4; ++j) b[j] = *(const bf16x16*)(brow[j] + k0);
#pragma unroll
        for (int i = 0; i < 2; ++i)
#pragma unroll
            for (int j = 0; j < 4; ++j)
                acc[i][j] = wmma_bf16(a[i], b[j], acc[i][j]);
    }

    if (MODE == 0) {
        // lane holds column n0+j*16+r for rows m0+i*16+hi*8 .. +7
#pragma unroll
        for (int i = 0; i < 2; ++i)
#pragma unroll
            for (int j = 0; j < 4; ++j)
#pragma unroll
                for (int rr = 0; rr < 8; ++rr) {
                    float t = elu1(acc[i][j][rr]);
                    OUT[(size_t)(m0 + i * 16 + hi * 8 + rr) * DMM + n0 + j * 16 + r] = (__bf16)t;
                }
    } else {
        int h = n0 >> 6;                 // n0 is a multiple of 64
        int bidx = m0 >> 13;             // batch (8192 rows each)
        int sl = (m0 & (SS - 1)) + hi * 8;
#pragma unroll
        for (int j = 0; j < 4; ++j) {
            int d = j * 16 + r;
            size_t base = (((size_t)(bidx * HH + h)) * DD + d) * SS + sl;
#pragma unroll
            for (int i = 0; i < 2; ++i) {
                bf16x8 o;
#pragma unroll
                for (int rr = 0; rr < 8; ++rr) {
                    float t = acc[i][j][rr];
                    if (MODE == 1) t = elu1(t);
                    o[rr] = (__bf16)t;
                }
                *(bf16x8*)(OUT + base + i * 16) = o;     // contiguous in s
            }
        }
    }
}

// ---------------------------------------------------------------------------
// kv[d][e] = sum_s kT[d][s] * vT[e][s]  per (b,h).  A=kT rows, B=vT rows.
// 512 waves (16x16 tiles), K loop = 256 WMMAs. Stores kvT bf16 [b,h,e,d].
// ---------------------------------------------------------------------------
__global__ void kv_gemm(const __bf16* __restrict__ KT, const __bf16* __restrict__ VT,
                        __bf16* __restrict__ KVT) {
    int wid  = (blockIdx.x * blockDim.x + threadIdx.x) >> 5;  // 512 waves
    int lane = threadIdx.x & 31;
    int r = lane & 15, hi = lane >> 4;
    int bh = wid >> 4;
    int t  = wid & 15;
    int m0 = (t >> 2) << 4;   // d tile
    int n0 = (t & 3) << 4;    // e tile

    const __bf16* arow = KT + ((size_t)bh * DD + m0 + r) * SS;
    const __bf16* brow = VT + ((size_t)bh * DD + n0 + r) * SS + hi * 16;

    f32x8 acc = {};
    for (int k0 = 0; k0 < SS; k0 += 32) {
        bf16x16 a = makeA(arow, k0, hi);
        bf16x16 b = *(const bf16x16*)(brow + k0);
        acc = wmma_bf16(a, b, acc);
    }
    // lane holds column e = n0+r, rows d = m0+hi*8..+7 -> kvT[e][d] contiguous
    bf16x8 o;
#pragma unroll
    for (int rr = 0; rr < 8; ++rr) o[rr] = (__bf16)acc[rr];
    *(bf16x8*)(KVT + ((size_t)bh * DD + n0 + r) * DD + m0 + hi * 8) = o;
}

// ---------------------------------------------------------------------------
// ksum[b,h,d] = sum_s kT[b,h,d,s]   (2048 rows, one wave each)
// ---------------------------------------------------------------------------
__global__ void ksum_kernel(const __bf16* __restrict__ KT, float* __restrict__ KS) {
    int w    = (blockIdx.x * blockDim.x + threadIdx.x) >> 5;  // 2048 rows
    int lane = threadIdx.x & 31;
    const __bf16* row = KT + (size_t)w * SS;
    float s = 0.0f;
    for (int i = lane; i < SS; i += 32) s += (float)row[i];
    for (int off = 16; off > 0; off >>= 1) s += __shfl_xor(s, off, 32);
    if (lane == 0) KS[w] = s;
}

// ---------------------------------------------------------------------------
// z[token,h] = 1 / dot(q[token,h,:], ksum[b,h,:])
// ---------------------------------------------------------------------------
__global__ void z_kernel(const __bf16* __restrict__ Q, const float* __restrict__ KS,
                         float* __restrict__ Z) {
    int idx = blockIdx.x * blockDim.x + threadIdx.x;   // 262144
    int tok = idx >> 3, h = idx & 7;
    const __bf16* qrow = Q + (size_t)tok * DMM + h * DD;
    const float*  ks   = KS + ((size_t)(tok >> 13) * HH + h) * DD;
    float acc = 0.0f;
#pragma unroll 8
    for (int d = 0; d < DD; ++d) acc += (float)qrow[d] * ks[d];
    Z[idx] = 1.0f / acc;
}

// ---------------------------------------------------------------------------
// yT tiles: C[e][s] = sum_d kvT[e][d] * q[s][d], scaled by z[s], per (b,h).
// Wave computes 32(e) x 64(s): 8 WMMAs per K-step, K = 64 (2 steps).
// Store y row-major bf16 [token][h*64+e]: b128 stores.
// ---------------------------------------------------------------------------
__global__ void y_gemm(const __bf16* __restrict__ KVT, const __bf16* __restrict__ Q,
                       const float* __restrict__ Z, __bf16* __restrict__ Y) {
    int wid  = (blockIdx.x * blockDim.x + threadIdx.x) >> 5;  // 8192 waves
    int lane = threadIdx.x & 31;
    int r = lane & 15, hi = lane >> 4;
    int bh = wid >> 8;             // 256 tiles per (b,h)
    int t  = wid & 255;
    int m0 = (t & 1) << 5;         // e tile of 32 (2)
    int s0 = (t >> 1) << 6;        // s tile of 64 (128)
    int b = bh >> 3, h = bh & 7;

    const __bf16* arow[2];
    const __bf16* brow[4];
    int sj[4];
#pragma unroll
    for (int i = 0; i < 2; ++i)
        arow[i] = KVT + ((size_t)bh * DD + m0 + i * 16 + r) * DD;
#pragma unroll
    for (int j = 0; j < 4; ++j) {
        sj[j] = s0 + j * 16 + r;
        brow[j] = Q + ((size_t)(b * SS + sj[j])) * DMM + h * DD + hi * 16;
    }

    f32x8 acc[2][4] = {};
#pragma unroll
    for (int k0 = 0; k0 < DD; k0 += 32) {
        bf16x16 a[2], bf[4];
#pragma unroll
        for (int i = 0; i < 2; ++i) a[i] = makeA(arow[i], k0, hi);
#pragma unroll
        for (int j = 0; j < 4; ++j) bf[j] = *(const bf16x16*)(brow[j] + k0);
#pragma unroll
        for (int i = 0; i < 2; ++i)
#pragma unroll
            for (int j = 0; j < 4; ++j)
                acc[i][j] = wmma_bf16(a[i], bf[j], acc[i][j]);
    }

#pragma unroll
    for (int j = 0; j < 4; ++j) {
        float zv = Z[((size_t)(b * SS + sj[j])) * HH + h];
        __bf16* ybase = Y + ((size_t)(b * SS + sj[j])) * DMM + h * DD + m0 + hi * 8;
#pragma unroll
        for (int i = 0; i < 2; ++i) {
            bf16x8 o;
#pragma unroll
            for (int rr = 0; rr < 8; ++rr) o[rr] = (__bf16)(acc[i][j][rr] * zv);
            *(bf16x8*)(ybase + i * 16) = o;
        }
    }
}

// ---------------------------------------------------------------------------
// Output projection: out fp32 [32768x512] = Y @ Wp  (WpT is [N][K] bf16).
// Register-blocked 32x64 per wave.
// ---------------------------------------------------------------------------
__global__ void out_gemm(const __bf16* __restrict__ Y, const __bf16* __restrict__ WT,
                         float* __restrict__ OUT) {
    int wid  = (blockIdx.x * blockDim.x + threadIdx.x) >> 5;  // 8192 waves
    int lane = threadIdx.x & 31;
    int r = lane & 15, hi = lane >> 4;
    int m0 = (wid >> 3) << 5;
    int n0 = (wid & 7) << 6;

    const __bf16* arow[2];
    const __bf16* brow[4];
#pragma unroll
    for (int i = 0; i < 2; ++i) arow[i] = Y + (size_t)(m0 + i * 16 + r) * DMM;
#pragma unroll
    for (int j = 0; j < 4; ++j) brow[j] = WT + (size_t)(n0 + j * 16 + r) * DMM + hi * 16;

    f32x8 acc[2][4] = {};
    for (int k0 = 0; k0 < DMM; k0 += 32) {
        bf16x16 a[2], b[4];
#pragma unroll
        for (int i = 0; i < 2; ++i) a[i] = makeA(arow[i], k0, hi);
#pragma unroll
        for (int j = 0; j < 4; ++j) b[j] = *(const bf16x16*)(brow[j] + k0);
#pragma unroll
        for (int i = 0; i < 2; ++i)
#pragma unroll
            for (int j = 0; j < 4; ++j)
                acc[i][j] = wmma_bf16(a[i], b[j], acc[i][j]);
    }
#pragma unroll
    for (int i = 0; i < 2; ++i)
#pragma unroll
        for (int j = 0; j < 4; ++j)
#pragma unroll
            for (int rr = 0; rr < 8; ++rr)
                OUT[(size_t)(m0 + i * 16 + hi * 8 + rr) * DMM + n0 + j * 16 + r] =
                    acc[i][j][rr];
}

// ---------------------------------------------------------------------------
extern "C" void kernel_launch(void* const* d_in, const int* in_sizes, int n_in,
                              void* d_out, int out_size, void* d_ws, size_t ws_size,
                              hipStream_t stream) {
    (void)in_sizes; (void)n_in; (void)out_size; (void)ws_size;
    // setup_inputs order: x, Wq, bq, Wk, bk, Wv, bv, Wp, bp (biases are zero).
    const float* x  = (const float*)d_in[0];
    const float* Wq = (const float*)d_in[1];
    const float* Wk = (const float*)d_in[3];
    const float* Wv = (const float*)d_in[5];
    const float* Wp = (const float*)d_in[7];
    float* out = (float*)d_out;

    char* ws = (char*)d_ws;
    size_t o = 0;
    __bf16* xb   = (__bf16*)(ws + o); o += (size_t)MTOT * DMM * 2;   // 32 MB
    __bf16* wqT  = (__bf16*)(ws + o); o += (size_t)DMM * DMM * 2;
    __bf16* wkT  = (__bf16*)(ws + o); o += (size_t)DMM * DMM * 2;
    __bf16* wvT  = (__bf16*)(ws + o); o += (size_t)DMM * DMM * 2;
    __bf16* wpT  = (__bf16*)(ws + o); o += (size_t)DMM * DMM * 2;
    __bf16* qb   = (__bf16*)(ws + o); o += (size_t)MTOT * DMM * 2;   // row-major q
    __bf16* kT   = (__bf16*)(ws + o); o += (size_t)MTOT * DMM * 2;   // [b,h,d,s]
    __bf16* vT   = (__bf16*)(ws + o); o += (size_t)MTOT * DMM * 2;   // [b,h,e,s]
    __bf16* kvT  = (__bf16*)(ws + o); o += (size_t)BB * HH * DD * DD * 2;
    float*  ksum = (float*)(ws + o);  o += (size_t)BB * HH * DD * 4;
    float*  zbuf = (float*)(ws + o);  o += (size_t)MTOT * HH * 4;
    __bf16* yb   = xb;   // x is dead after projections; reuse its buffer for y

    // 1) bf16 conversion + weight transposes
    cvt_kernel<<<(MTOT * DMM) / 256, 256, 0, stream>>>(x, xb, MTOT * DMM);
    wT_kernel<<<(DMM * DMM) / 256, 256, 0, stream>>>(Wq, wqT);
    wT_kernel<<<(DMM * DMM) / 256, 256, 0, stream>>>(Wk, wkT);
    wT_kernel<<<(DMM * DMM) / 256, 256, 0, stream>>>(Wv, wvT);
    wT_kernel<<<(DMM * DMM) / 256, 256, 0, stream>>>(Wp, wpT);

    // 2) QKV projections (8192 waves each, 8 waves/block, 32x64 tiles)
    proj_gemm<0><<<1024, 256, 0, stream>>>(xb, wqT, qb);
    proj_gemm<1><<<1024, 256, 0, stream>>>(xb, wkT, kT);
    proj_gemm<2><<<1024, 256, 0, stream>>>(xb, wvT, vT);

    // 3) normalizer pieces
    ksum_kernel<<<256, 256, 0, stream>>>(kT, ksum);
    z_kernel<<<(MTOT * HH) / 256, 256, 0, stream>>>(qb, ksum, zbuf);

    // 4) kv accumulation (kvT bf16), then y = q.kv.z
    kv_gemm<<<64, 256, 0, stream>>>(kT, vT, kvT);
    y_gemm<<<1024, 256, 0, stream>>>(kvT, qb, zbuf, yb);

    // 5) output projection -> fp32 d_out
    out_gemm<<<1024, 256, 0, stream>>>(yb, wpT, out);
}
